// Decoder_22368189678032
// MI455X (gfx1250) — matrix operand order repaired and checked by reading, not compiled, not run on hardware
//
#include <hip/hip_runtime.h>
#include <hip/hip_bf16.h>
#include <math.h>

// ---- problem constants ----
#define NSLOTS   30
#define HIDDEN   768
#define NL       200     // num labels
#define NTOK     122880
#define ROWS     16      // rows (tokens of one slot) per block
#define XSTRB    776     // padded bf16 LDS row stride (768+8): 1552B % 256B = 16
#define LSTR     208     // padded class dim (13 tiles of 16)
#define NTILES   13
#define KSTEPS   24      // 768 / 32
#define MBLOCKS  256     // 4096 rows-per-slot / ROWS
#define NBLK     (MBLOCKS * NSLOTS)

typedef __attribute__((ext_vector_type(16))) __bf16 v16bf;
typedef __attribute__((ext_vector_type(8)))  __bf16 v8bf;
typedef __attribute__((ext_vector_type(4)))  __bf16 v4bf;
typedef __attribute__((ext_vector_type(8)))  float  v8f;

// ---------------------------------------------------------------------------
// Prep: W[s][d][c] (f32) -> blocked transposed bf16 hi/lo:
//   Wt[s][kb][c][dk], kb = d/32, dk = d%32, c padded to 208 (zeros past 200).
// Each lane's WMMA B-fragment is then a contiguous 32B load.
// grid = (24, 30), block = 256
// ---------------------------------------------------------------------------
__global__ __launch_bounds__(256) void prep_w_kernel(
    const float* __restrict__ W, __bf16* __restrict__ wh, __bf16* __restrict__ wl)
{
    __shared__ float wt[32 * NL];
    const int tid = threadIdx.x;
    const int kb  = blockIdx.x;
    const int s   = blockIdx.y;
    // W region for (s, kb): 32 rows of 200 floats, contiguous in memory.
    const size_t base = ((size_t)s * HIDDEN + (size_t)kb * 32) * NL;
    for (int i = tid; i < 32 * NL; i += 256) wt[i] = W[base + i];
    __syncthreads();

    const int c = tid;
    if (c < LSTR) {
        v16bf h0, h1, l0, l1;
        #pragma unroll
        for (int j = 0; j < 32; ++j) {
            float w = (c < NL) ? wt[j * NL + c] : 0.0f;
            __bf16 h = (__bf16)w;
            __bf16 l = (__bf16)(w - (float)h);
            if (j < 16) { h0[j] = h; l0[j] = l; }
            else        { h1[j - 16] = h; l1[j - 16] = l; }
        }
        const size_t off = (((size_t)s * KSTEPS + kb) * LSTR + c) * 32;
        *(v16bf*)(wh + off)      = h0;
        *(v16bf*)(wh + off + 16) = h1;
        *(v16bf*)(wl + off)      = l0;
        *(v16bf*)(wl + off + 16) = l1;
    }
}

// Build a 16-element A fragment from two contiguous 8-element LDS loads.
__device__ __forceinline__ v16bf load_afrag(const __bf16* row, int off1, int off2) {
    const v8bf a0 = *(const v8bf*)(row + off1);
    const v8bf a1 = *(const v8bf*)(row + off2);
    return __builtin_shufflevector(a0, a1, 0, 1, 2, 3, 4, 5, 6, 7,
                                           8, 9, 10, 11, 12, 13, 14, 15);
}

// ---------------------------------------------------------------------------
// Main: per (slot, 16-row block): GEMM via split-bf16 WMMA, then log-softmax,
// NLL and argmax. grid = (256, 30), block = 256 (8 wave32s).
// ---------------------------------------------------------------------------
__global__ __launch_bounds__(256) void head_loss_kernel(
    const float* __restrict__ X, const int* __restrict__ ids,
    const float* __restrict__ bias,
    const __bf16* __restrict__ wh, const __bf16* __restrict__ wl,
    float* __restrict__ preds_out, float* __restrict__ part, float* __restrict__ cnt)
{
    __shared__ __bf16 xh[ROWS * XSTRB];  // staged x rows, bf16 hi
    __shared__ __bf16 xl[ROWS * XSTRB];  // staged x rows, bf16 residual
    __shared__ float  lg[ROWS * LSTR];   // logits
    __shared__ float  rn[ROWS];
    __shared__ float  rc[ROWS];

    const int tid = threadIdx.x;
    const int s   = blockIdx.y;
    const int b0  = blockIdx.x * ROWS;   // row (within-slot) base

    // Stage 16 rows of x, splitting f32 -> bf16 hi + residual lo ONCE per block.
    // Token t = (b0+r)*30 + s; coalesced float4 loads, 8B bf16x4 LDS stores.
    for (int f = tid; f < ROWS * (HIDDEN / 4); f += 256) {
        const int r  = f / (HIDDEN / 4);
        const int c4 = f % (HIDDEN / 4);
        const float4 v = *(const float4*)(X + (size_t)((b0 + r) * NSLOTS + s) * HIDDEN + c4 * 4);
        v4bf hv, lv;
        const float xv[4] = {v.x, v.y, v.z, v.w};
        #pragma unroll
        for (int j = 0; j < 4; ++j) {
            const __bf16 h = (__bf16)xv[j];
            hv[j] = h;
            lv[j] = (__bf16)(xv[j] - (float)h);
        }
        *(v4bf*)(&xh[r * XSTRB + c4 * 4]) = hv;
        *(v4bf*)(&xl[r * XSTRB + c4 * 4]) = lv;
    }
    __syncthreads();

    const int lane = tid & 31;
    const int wid  = tid >> 5;
    const int lm   = lane & 15;
    const int g    = lane >> 4;

    for (int tile = wid; tile < NTILES; tile += 8) {   // wave-uniform loop
        const int c0 = tile * 16;
        const int c  = c0 + lm;
        v8f acc0 = {}, acc1 = {}, acc2 = {};
        const __bf16* ph = wh + (((size_t)s * KSTEPS) * LSTR + c) * 32 + 16 * g;
        const __bf16* pl = wl + (((size_t)s * KSTEPS) * LSTR + c) * 32 + 16 * g;
        const __bf16* ahrow = &xh[lm * XSTRB];
        const __bf16* alrow = &xl[lm * XSTRB];

        #pragma unroll 2
        for (int kb = 0; kb < KSTEPS; ++kb) {
            const int k0 = kb * 32;
            // A fragment (contiguous k order, matching B): element e -> k0+16g+e;
            // contiguous 8-element runs -> ds_load_b128.
            const int o1 = k0 + 16 * g;
            const int o2 = k0 + 16 * g + 8;
            const v16bf ah = load_afrag(ahrow, o1, o2);
            const v16bf al = load_afrag(alrow, o1, o2);
            // B fragments: 32B contiguous per lane from blocked Wt.
            const v16bf bh = *(const v16bf*)(ph + (size_t)kb * (LSTR * 32));
            const v16bf bl = *(const v16bf*)(pl + (size_t)kb * (LSTR * 32));
            // Three independent accumulation chains: hi*hi + lo*hi + hi*lo.
            acc0 = __builtin_amdgcn_wmma_f32_16x16x32_bf16(false, ah, false, bh, (short)0, acc0, false, false);
            acc1 = __builtin_amdgcn_wmma_f32_16x16x32_bf16(false, al, false, bh, (short)0, acc1, false, false);
            acc2 = __builtin_amdgcn_wmma_f32_16x16x32_bf16(false, ah, false, bl, (short)0, acc2, false, false);
        }
        if (c < NL) {
            const float bv = bias[s * NL + c];
            // C/D layout: lane col = c, VGPR v row = v + 8*g.
            #pragma unroll
            for (int v = 0; v < 8; ++v)
                lg[(v + 8 * g) * LSTR + c] = acc0[v] + acc1[v] + acc2[v] + bv;
        }
    }
    __syncthreads();

    // Per-row log-softmax + NLL + argmax (tiny vs GEMM).
    if (tid < ROWS) {
        const int t = (b0 + tid) * NSLOTS + s;
        const float* row = &lg[tid * LSTR];
        float mx = row[0]; int am = 0;
        for (int c2 = 1; c2 < NL; ++c2) {
            const float v = row[c2];
            if (v > mx) { mx = v; am = c2; }
        }
        float se = 0.0f;
        for (int c2 = 0; c2 < NL; ++c2) se += expf(row[c2] - mx);
        const int id = ids[t];
        const bool valid = (id != -1);
        float nll = 0.0f;
        if (valid) nll = (logf(se) + mx) - row[id];
        preds_out[t] = valid ? (float)am : -1.0f;
        rn[tid] = nll;
        rc[tid] = valid ? 1.0f : 0.0f;
    }
    __syncthreads();
    if (tid == 0) {
        float a = 0.0f, b = 0.0f;
        #pragma unroll
        for (int i = 0; i < ROWS; ++i) { a += rn[i]; b += rc[i]; }
        const int bid = blockIdx.y * gridDim.x + blockIdx.x;
        part[bid] = a; cnt[bid] = b;
    }
}

// ---------------------------------------------------------------------------
// Deterministic reduction of per-block partials -> loss.
// ---------------------------------------------------------------------------
__global__ __launch_bounds__(256) void finalize_kernel(
    const float* __restrict__ part, const float* __restrict__ cnt,
    float* __restrict__ out, int n)
{
    __shared__ float ss[256], sc[256];
    const int tid = threadIdx.x;
    float a = 0.0f, b = 0.0f;
    for (int i = tid; i < n; i += 256) { a += part[i]; b += cnt[i]; }
    ss[tid] = a; sc[tid] = b;
    __syncthreads();
    for (int st = 128; st > 0; st >>= 1) {
        if (tid < st) { ss[tid] += ss[tid + st]; sc[tid] += sc[tid + st]; }
        __syncthreads();
    }
    if (tid == 0) out[0] = (sc[0] > 0.0f) ? (ss[0] / sc[0]) : 0.0f;
}

extern "C" void kernel_launch(void* const* d_in, const int* in_sizes, int n_in,
                              void* d_out, int out_size, void* d_ws, size_t ws_size,
                              hipStream_t stream) {
    const float* X    = (const float*)d_in[0];   // [122880, 768] f32
    /* d_in[1] = op_ids, unused by the reference */
    const int*   ids  = (const int*)d_in[2];     // [122880] int32
    const float* W    = (const float*)d_in[3];   // [30, 768, 200] f32
    const float* bias = (const float*)d_in[4];   // [30, 200] f32
    float* out = (float*)d_out;                  // [0]=loss, [1..122880]=preds

    // Workspace layout.
    char* ws = (char*)d_ws;
    float* part = (float*)(ws + 256);
    float* cnt  = (float*)(ws + 256 + (size_t)NBLK * 4);
    const size_t off_wt = 256 + 2 * (size_t)NBLK * 4;          // 61696, 256-aligned
    __bf16* wh = (__bf16*)(ws + off_wt);
    __bf16* wl = wh + (size_t)NSLOTS * KSTEPS * LSTR * 32;     // ~9.6 MB each

    prep_w_kernel<<<dim3(KSTEPS, NSLOTS), 256, 0, stream>>>(W, wh, wl);
    head_loss_kernel<<<dim3(MBLOCKS, NSLOTS), 256, 0, stream>>>(
        X, ids, bias, wh, wl, out + 1, part, cnt);
    finalize_kernel<<<1, 256, 0, stream>>>(part, cnt, out, NBLK);
}
